// GGUFLayer_88802743812109
// MI455X (gfx1250) — matrix operand order repaired and checked by reading, not compiled, not run on hardware
//
#include <hip/hip_runtime.h>

// ---------------------------------------------------------------------------
// Q4_0 dequant + GEMM for MI455X (gfx1250, wave32, WMMA)
//   out[n,o] = sum_k x[n,k] * deq(qweight,scales)[o,k] + bias[o]
//   M=4096, K=8192, O=8192, Q4_0 block=32
//
// Pipeline (when workspace permits, 192 MB):
//   1) convert_x : x f32 -> bf16                (ws, 64 MB)
//   2) dequant_w : Q4_0  -> bf16 W[O][K]        (ws, 128 MB; L2-resident after)
//   3) gemm_bf16 : double-buffered LDS tiles staged with
//                  GLOBAL_LOAD_ASYNC_TO_LDS_B128 (ASYNCcnt), WMMA bf16 compute
// Fallback: fused dequant+GEMM (round-1 kernel).
// ---------------------------------------------------------------------------

typedef __attribute__((ext_vector_type(16))) __bf16        v16bf;
typedef __attribute__((ext_vector_type(8)))  float         v8f;
typedef __attribute__((ext_vector_type(8)))  unsigned int  v8u;
typedef int v4i __attribute__((vector_size(16)));          // matches builtin param

static constexpr int Mdim = 4096;
static constexpr int Kdim = 8192;
static constexpr int Odim = 8192;
static constexpr int NBlk = Kdim / 32;   // 256 Q4_0 blocks per weight row
static constexpr int BM   = 128;         // M tile
static constexpr int BN   = 128;         // O tile
static constexpr int BK   = 64;          // K chunk (bf16 elems)
static constexpr int LDT  = 72;          // LDS row stride (bf16 elems), padded
static constexpr int NCHUNK = Kdim / BK; // 128

// -------------------- helpers ----------------------------------------------

// float -> bf16 (round-to-nearest-even), raw u16 bits
__device__ __forceinline__ unsigned short f2bf(float f) {
    unsigned u = __float_as_uint(f);
    u += 0x7FFFu + ((u >> 16) & 1u);
    return (unsigned short)(u >> 16);
}
__device__ __forceinline__ unsigned pack2(float a, float b) {
    return (unsigned)f2bf(a) | ((unsigned)f2bf(b) << 16);
}

// Q4_0 block (16 byte-valued int32 + scale) -> 32 bf16 packed in 4 uint4
// low nibble of byte j -> elem j ; high nibble -> elem 16+j
__device__ __forceinline__ void dequant_block(int4 q0, int4 q1, int4 q2, int4 q3,
                                              float s, uint4& w0, uint4& w1,
                                              uint4& w2, uint4& w3) {
    int qb[16] = {q0.x, q0.y, q0.z, q0.w, q1.x, q1.y, q1.z, q1.w,
                  q2.x, q2.y, q2.z, q2.w, q3.x, q3.y, q3.z, q3.w};
    unsigned lopk[8], hipk[8];
    #pragma unroll
    for (int t = 0; t < 8; ++t) {
        int b0v = qb[2 * t], b1v = qb[2 * t + 1];
        lopk[t] = pack2((float)((b0v & 0xF) - 8) * s,
                        (float)((b1v & 0xF) - 8) * s);
        hipk[t] = pack2((float)(((b0v >> 4) & 0xF) - 8) * s,
                        (float)(((b1v >> 4) & 0xF) - 8) * s);
    }
    w0 = make_uint4(lopk[0], lopk[1], lopk[2], lopk[3]);
    w1 = make_uint4(lopk[4], lopk[5], lopk[6], lopk[7]);
    w2 = make_uint4(hipk[0], hipk[1], hipk[2], hipk[3]);
    w3 = make_uint4(hipk[4], hipk[5], hipk[6], hipk[7]);
}

// -------------------- async global->LDS copy (CDNA5) ------------------------

#if defined(__AMDGCN__) && __has_builtin(__builtin_amdgcn_global_load_async_to_lds_b128)
#define USE_ASYNC 1
#else
#define USE_ASYNC 0
#endif

__device__ __forceinline__ void cp16_g2l(const unsigned short* g, unsigned short* l) {
#if USE_ASYNC
    __builtin_amdgcn_global_load_async_to_lds_b128(
        (__attribute__((address_space(1))) v4i*)g,
        (__attribute__((address_space(3))) v4i*)l,
        0, 0);
#else
    *(uint4*)l = *(const uint4*)g;
#endif
}

template <int N>
__device__ __forceinline__ void wait_async() {
#if USE_ASYNC
#if __has_builtin(__builtin_amdgcn_s_wait_asynccnt)
    __builtin_amdgcn_s_wait_asynccnt(N);
#else
    asm volatile("s_wait_asynccnt %0" ::"i"(N) : "memory");
#endif
#endif
}

// -------------------- kernel 1: x f32 -> bf16 --------------------------------

__global__ __launch_bounds__(256)
void convert_x_bf16(const float* __restrict__ x, unsigned short* __restrict__ xa) {
    size_t i = ((size_t)blockIdx.x * 256 + threadIdx.x) * 8;
    const float4* p = (const float4*)(x + i);
    float4 a = p[0], b = p[1];
    uint4 w = make_uint4(pack2(a.x, a.y), pack2(a.z, a.w),
                         pack2(b.x, b.y), pack2(b.z, b.w));
    *(uint4*)(xa + i) = w;
}

// -------------------- kernel 2: Q4_0 -> bf16 W[O][K] -------------------------

__global__ __launch_bounds__(256)
void dequant_w_bf16(const int* __restrict__ qw, const float* __restrict__ scales,
                    unsigned short* __restrict__ wb) {
    size_t idx = (size_t)blockIdx.x * 256 + threadIdx.x;   // o*NBlk + nb
    const int4* qp = (const int4*)(qw + idx * 16);
    int4 q0 = qp[0], q1 = qp[1], q2 = qp[2], q3 = qp[3];
    float s = scales[idx];
    uint4 w0, w1, w2, w3;
    dequant_block(q0, q1, q2, q3, s, w0, w1, w2, w3);
    size_t o  = idx / NBlk;
    size_t nb = idx % NBlk;
    uint4* dst = (uint4*)(wb + o * Kdim + nb * 32);
    dst[0] = w0; dst[1] = w1; dst[2] = w2; dst[3] = w3;
}

// -------------------- kernel 3: bf16 GEMM with async LDS pipeline ------------

__global__ __launch_bounds__(256)
void gemm_bf16_wmma(const unsigned short* __restrict__ A,  // [M][K] bf16 bits
                    const unsigned short* __restrict__ B,  // [O][K] bf16 bits
                    const float* __restrict__ bias,
                    float*       __restrict__ out)
{
    __shared__ __align__(16) unsigned short lds_a[2][BM * LDT];
    __shared__ __align__(16) unsigned short lds_b[2][BN * LDT];

    const int tid   = threadIdx.x;
    const int lane  = tid & 31;
    const int wave  = tid >> 5;
    const int wm    = (wave & 3) * 32;   // wave M offset in tile
    const int wn    = (wave >> 2) * 64;  // wave N offset in tile
    const int tileM = blockIdx.y * BM;
    const int tileN = blockIdx.x * BN;

    v8f acc[2][4];
    #pragma unroll
    for (int i = 0; i < 2; ++i)
        #pragma unroll
        for (int j = 0; j < 4; ++j)
            #pragma unroll
            for (int e = 0; e < 8; ++e)
                acc[i][j][e] = 0.0f;

    // staging map: 1024 pieces of 16B per tile side; 4 pieces per thread
    const unsigned short* ga[4];
    const unsigned short* gb[4];
    int loff[4];
    #pragma unroll
    for (int p = 0; p < 4; ++p) {
        int lin = tid + p * 256;        // 0..1023
        int row = lin >> 3;             // 0..127
        int c8  = (lin & 7) * 8;        // 0..56 (bf16 elems)
        ga[p]   = A + (size_t)(tileM + row) * Kdim + c8;
        gb[p]   = B + (size_t)(tileN + row) * Kdim + c8;
        loff[p] = row * LDT + c8;
    }

    // prologue: issue chunk 0 into buffer 0
    #pragma unroll
    for (int p = 0; p < 4; ++p) {
        cp16_g2l(ga[p], &lds_a[0][loff[p]]);
        cp16_g2l(gb[p], &lds_b[0][loff[p]]);
    }

    for (int kc = 0; kc < NCHUNK; ++kc) {
        const int cur = kc & 1;
        if (kc + 1 < NCHUNK) {
            const int nxt = cur ^ 1;
            const int ko  = (kc + 1) * BK;
            #pragma unroll
            for (int p = 0; p < 4; ++p) {
                cp16_g2l(ga[p] + ko, &lds_a[nxt][loff[p]]);
                cp16_g2l(gb[p] + ko, &lds_b[nxt][loff[p]]);
            }
            wait_async<8>();   // chunk kc's 8 async ops have completed
        } else {
            wait_async<0>();
        }
        __syncthreads();       // all waves' tile data visible

        const unsigned short* abase = lds_a[cur];
        const unsigned short* bbase = lds_b[cur];

        #pragma unroll
        for (int ks = 0; ks < 2; ++ks) {
            const int k0 = ks * 32;

            // A fragments (lane = M row; K halves split across lane groups)
            v16bf afrag[2];
            #pragma unroll
            for (int mi = 0; mi < 2; ++mi) {
                int am = wm + mi * 16 + (lane & 15);
                int ak = k0 + ((lane & 16) ? 16 : 0);
                uint4 a0 = *(const uint4*)&abase[am * LDT + ak];       // K+0..7
                uint4 a1 = *(const uint4*)&abase[am * LDT + ak + 8];   // K+8..15
                v8u av = {a0.x, a0.y, a0.z, a0.w, a1.x, a1.y, a1.z, a1.w};
                afrag[mi] = __builtin_bit_cast(v16bf, av);
            }
            // B fragments (lane = N col; lanes 0-15: K 0..15, 16-31: K 16..31)
            v16bf bfrag[4];
            #pragma unroll
            for (int ni = 0; ni < 4; ++ni) {
                int bn = wn + ni * 16 + (lane & 15);
                int bk = k0 + ((lane & 16) ? 16 : 0);
                const uint4* pb = (const uint4*)&bbase[bn * LDT + bk];
                uint4 b0v = pb[0], b1v = pb[1];
                v8u bv = {b0v.x, b0v.y, b0v.z, b0v.w, b1v.x, b1v.y, b1v.z, b1v.w};
                bfrag[ni] = __builtin_bit_cast(v16bf, bv);
            }

            #pragma unroll
            for (int mi = 0; mi < 2; ++mi)
                #pragma unroll
                for (int ni = 0; ni < 4; ++ni)
                    acc[mi][ni] = __builtin_amdgcn_wmma_f32_16x16x32_bf16(
                        false, afrag[mi], false, bfrag[ni],
                        (short)0, acc[mi][ni], false, false);
        }
        __syncthreads();       // tile reads done; buffer may be overwritten
    }

    // epilogue: bias add + store f32 (16-lane groups store 64B rows)
    #pragma unroll
    for (int mi = 0; mi < 2; ++mi) {
        #pragma unroll
        for (int ni = 0; ni < 4; ++ni) {
            const int col = tileN + wn + ni * 16 + (lane & 15);
            const int row = tileM + wm + mi * 16 + ((lane & 16) ? 8 : 0);
            const float bv = bias[col];
            #pragma unroll
            for (int r = 0; r < 8; ++r)
                out[(size_t)(row + r) * Odim + col] = acc[mi][ni][r] + bv;
        }
    }
}

// -------------------- fallback: fused dequant + GEMM (round-1) ---------------

__global__ __launch_bounds__(256)
void q4_0_gemm_fused(const float* __restrict__ x,
                     const int*   __restrict__ qw,
                     const float* __restrict__ scales,
                     const float* __restrict__ bias,
                     float*       __restrict__ out)
{
    __shared__ __align__(16) unsigned short lds_a[BM * LDT];
    __shared__ __align__(16) unsigned short lds_b[BN * LDT];

    const int tid   = threadIdx.x;
    const int lane  = tid & 31;
    const int wave  = tid >> 5;
    const int wm    = (wave & 3) * 32;
    const int wn    = (wave >> 2) * 64;
    const int tileM = blockIdx.y * BM;
    const int tileN = blockIdx.x * BN;

    v8f acc[2][4];
    #pragma unroll
    for (int i = 0; i < 2; ++i)
        #pragma unroll
        for (int j = 0; j < 4; ++j)
            #pragma unroll
            for (int e = 0; e < 8; ++e)
                acc[i][j][e] = 0.0f;

    const int bo   = tid >> 1;
    const int bblk = tid & 1;
    const int og   = tileN + bo;

    for (int kk = 0; kk < Kdim; kk += BK) {
        float4 ax[4][2];
        int arow[4], acol[4];
        #pragma unroll
        for (int p = 0; p < 4; ++p) {
            int lin  = tid + p * 256;
            arow[p]  = lin >> 3;
            acol[p]  = (lin & 7) * 8;
            const float4* px =
                (const float4*)(x + (size_t)(tileM + arow[p]) * Kdim + kk + acol[p]);
            ax[p][0] = px[0];
            ax[p][1] = px[1];
        }
        const int nb0 = (kk >> 5) + bblk;
        const int4* qp = (const int4*)(qw + ((size_t)og * NBlk + nb0) * 16);
        int4 q0 = qp[0], q1 = qp[1], q2 = qp[2], q3 = qp[3];
        const float s = scales[(size_t)og * NBlk + nb0];

        __syncthreads();

        #pragma unroll
        for (int p = 0; p < 4; ++p) {
            uint4 w = make_uint4(pack2(ax[p][0].x, ax[p][0].y),
                                 pack2(ax[p][0].z, ax[p][0].w),
                                 pack2(ax[p][1].x, ax[p][1].y),
                                 pack2(ax[p][1].z, ax[p][1].w));
            *(uint4*)&lds_a[arow[p] * LDT + acol[p]] = w;
        }
        {
            uint4 w0, w1, w2, w3;
            dequant_block(q0, q1, q2, q3, s, w0, w1, w2, w3);
            uint4* dst = (uint4*)&lds_b[bo * LDT + bblk * 32];
            dst[0] = w0; dst[1] = w1; dst[2] = w2; dst[3] = w3;
        }

        __syncthreads();

        #pragma unroll
        for (int ks = 0; ks < 2; ++ks) {
            const int k0 = ks * 32;
            v16bf afrag[2];
            #pragma unroll
            for (int mi = 0; mi < 2; ++mi) {
                int am = wm + mi * 16 + (lane & 15);
                int ak = k0 + ((lane & 16) ? 16 : 0);
                uint4 a0 = *(const uint4*)&lds_a[am * LDT + ak];
                uint4 a1 = *(const uint4*)&lds_a[am * LDT + ak + 8];
                v8u av = {a0.x, a0.y, a0.z, a0.w, a1.x, a1.y, a1.z, a1.w};
                afrag[mi] = __builtin_bit_cast(v16bf, av);
            }
            v16bf bfrag[4];
            #pragma unroll
            for (int ni = 0; ni < 4; ++ni) {
                int bn = wn + ni * 16 + (lane & 15);
                int bk = k0 + ((lane & 16) ? 16 : 0);
                const uint4* pb = (const uint4*)&lds_b[bn * LDT + bk];
                uint4 b0v = pb[0], b1v = pb[1];
                v8u bv = {b0v.x, b0v.y, b0v.z, b0v.w, b1v.x, b1v.y, b1v.z, b1v.w};
                bfrag[ni] = __builtin_bit_cast(v16bf, bv);
            }
            #pragma unroll
            for (int mi = 0; mi < 2; ++mi)
                #pragma unroll
                for (int ni = 0; ni < 4; ++ni)
                    acc[mi][ni] = __builtin_amdgcn_wmma_f32_16x16x32_bf16(
                        false, afrag[mi], false, bfrag[ni],
                        (short)0, acc[mi][ni], false, false);
        }
    }

    #pragma unroll
    for (int mi = 0; mi < 2; ++mi) {
        #pragma unroll
        for (int ni = 0; ni < 4; ++ni) {
            const int col = tileN + wn + ni * 16 + (lane & 15);
            const int row = tileM + wm + mi * 16 + ((lane & 16) ? 8 : 0);
            const float bv = bias[col];
            #pragma unroll
            for (int r = 0; r < 8; ++r)
                out[(size_t)(row + r) * Odim + col] = acc[mi][ni][r] + bv;
        }
    }
}

// -------------------- launch -------------------------------------------------

extern "C" void kernel_launch(void* const* d_in, const int* in_sizes, int n_in,
                              void* d_out, int out_size, void* d_ws, size_t ws_size,
                              hipStream_t stream) {
    (void)in_sizes; (void)n_in; (void)out_size;
    const float* x      = (const float*)d_in[0];
    const int*   qw     = (const int*)d_in[1];
    const float* scales = (const float*)d_in[2];
    const float* bias   = (const float*)d_in[3];
    float*       out    = (float*)d_out;

    const size_t needA = (size_t)Mdim * Kdim * 2;   //  64 MB bf16 x
    const size_t needB = (size_t)Odim * Kdim * 2;   // 128 MB bf16 W

    dim3 grid(Odim / BN, Mdim / BM);   // (64, 32)
    dim3 block(256);

    if (ws_size >= needA + needB) {
        unsigned short* xa = (unsigned short*)d_ws;
        unsigned short* wb = (unsigned short*)((char*)d_ws + needA);
        convert_x_bf16<<<(unsigned)(((size_t)Mdim * Kdim) / (8 * 256)), block, 0, stream>>>(x, xa);
        dequant_w_bf16<<<(unsigned)(((size_t)Odim * NBlk) / 256), block, 0, stream>>>(qw, scales, wb);
        gemm_bf16_wmma<<<grid, block, 0, stream>>>(xa, wb, bias, out);
    } else {
        q4_0_gemm_fused<<<grid, block, 0, stream>>>(x, qw, scales, bias, out);
    }
}
